// MoELayer_33655363731932
// MI455X (gfx1250) — matrix operand order repaired and checked by reading, not compiled, not run on hardware
//
#include <hip/hip_runtime.h>
#include <hip/hip_bf16.h>

typedef _Float16 f16;
typedef _Float16 v16h __attribute__((ext_vector_type(16)));
typedef _Float16 v8h  __attribute__((ext_vector_type(8)));
typedef float    v8f  __attribute__((ext_vector_type(8)));

#define B_   16
#define C_   64
#define HW_  16384
#define E_   8
#define NT   128      // spatial columns per block
#define LDP  72       // padded LDS row (f16 elems) for activation tiles

// gw * gelu_tanh(t)  with  hs = 0.5f * gw  pre-folded
__device__ __forceinline__ float gelu_scaled(float t, float hs) {
    const float c0 = 0.7978845608028654f;   // sqrt(2/pi)
    const float c1 = 0.044715f;
    float t3 = t * t * t;
    return hs * t * (1.0f + tanhf(c0 * (t + c1 * t3)));
}

// ---------------- Phase 1: global average pool -> pooled[B,C] ----------------
__global__ __launch_bounds__(256)
void pool_kernel(const float* __restrict__ x, float* __restrict__ pooled) {
    __shared__ float red[256];
    const int row = blockIdx.x;                 // b*C + c
    const int tid = threadIdx.x;
    const float* p = x + (size_t)row * HW_;
    float s = 0.0f;
    #pragma unroll
    for (int i = 0; i < 16; ++i) {
        float4 v = *(const float4*)(p + (((i << 8) + tid) << 2));
        s += v.x + v.y + v.z + v.w;
    }
    red[tid] = s;
    __syncthreads();
    for (int o = 128; o > 0; o >>= 1) {
        if (tid < o) red[tid] += red[tid + o];
        __syncthreads();
    }
    if (tid == 0) pooled[row] = red[0] * (1.0f / (float)HW_);
}

// --- Phase 2: weight f32->f16 conversion into WMMA-A fragment order + gate ---
// Fragment layout per expert (4096 f16): frag index (kk,hi,m) -> 16 contiguous
// f16 = W[m][kk*32+hi*16 + {0..15}]
__global__ __launch_bounds__(256)
void gate_kernel(const float* __restrict__ pooled,
                 const float* __restrict__ Wg, const float* __restrict__ bg,
                 const float* __restrict__ W1, const float* __restrict__ W2,
                 f16* __restrict__ W1h, f16* __restrict__ W2h,
                 int* __restrict__ sel_idx, float* __restrict__ sel_w) {
    const int tid = threadIdx.x;
    for (int d = tid; d < E_ * C_ * C_; d += 256) {
        int e  = d >> 12;
        int t  = d & 4095;
        int q  = t & 15;
        int m  = (t >> 4) & 63;
        int hi = (t >> 10) & 1;
        int kk = t >> 11;
        int k  = kk * 32 + hi * 16 + q;
        int src = (e << 12) + (m << 6) + k;
        W1h[d] = (f16)W1[src];
        W2h[d] = (f16)W2[src];
    }
    if (tid < B_) {
        const int b = tid;
        float logit[E_];
        #pragma unroll
        for (int e = 0; e < E_; ++e) logit[e] = bg[e];
        for (int c = 0; c < C_; ++c) {
            float pv = pooled[b * C_ + c];
            #pragma unroll
            for (int e = 0; e < E_; ++e) logit[e] += pv * Wg[c * E_ + e];
        }
        float mx = logit[0];
        #pragma unroll
        for (int e = 1; e < E_; ++e) mx = fmaxf(mx, logit[e]);
        float w[E_]; float den = 0.0f;
        #pragma unroll
        for (int e = 0; e < E_; ++e) { w[e] = expf(logit[e] - mx); den += w[e]; }
        float inv = 1.0f / den;
        #pragma unroll
        for (int e = 0; e < E_; ++e) w[e] *= inv;
        int i0 = 0;
        #pragma unroll
        for (int e = 1; e < E_; ++e) if (w[e] > w[i0]) i0 = e;   // first max (jax tiebreak)
        int i1 = (i0 == 0) ? 1 : 0;
        #pragma unroll
        for (int e = 0; e < E_; ++e) if (e != i0 && w[e] > w[i1]) i1 = e;
        sel_idx[b * 2 + 0] = i0;  sel_idx[b * 2 + 1] = i1;
        sel_w [b * 2 + 0] = w[i0]; sel_w [b * 2 + 1] = w[i1];
    }
}

// ---- Phase 3: per (batch, spatial tile): out = x + sum_{top2} w_e * y_e ----
__global__ __launch_bounds__(256)
void moe_expert_kernel(const float* __restrict__ x,
                       const float* __restrict__ kvec,
                       const float* __restrict__ b1g,
                       const float* __restrict__ b2g,
                       const f16*  __restrict__ W1h,
                       const f16*  __restrict__ W2h,
                       const int*  __restrict__ sel_idx,
                       const float* __restrict__ sel_w,
                       float* __restrict__ out) {
    __shared__ f16  xs[NT * LDP];     // x tile, f16, [n][c]  (B-operand layout)
    __shared__ f16  hs[NT * LDP];     // gw-scaled gelu output tile, f16, [n][c]
    __shared__ f16  w1s[C_ * C_];     // expert W1 fragments (pre-swizzled)
    __shared__ f16  w2s[C_ * C_];
    __shared__ float ks[C_], b1s[C_], b2s[C_];   // k, b1*k, gw*b2

    const int tid = threadIdx.x;
    const int b   = blockIdx.x >> 7;          // 128 tiles per batch image
    const int hw0 = (blockIdx.x & 127) * NT;
    const float* xb = x + (size_t)b * C_ * HW_;

    // ---- stage x tile: f32 global -> f16 LDS transposed [n][c] ----
    #pragma unroll
    for (int i = 0; i < 8; ++i) {
        int f  = (i << 8) + tid;              // float4 index; 32 per channel
        int c  = f >> 5;
        int n4 = (f & 31) << 2;
        float4 v = *(const float4*)(xb + (size_t)c * HW_ + hw0 + n4);
        xs[(n4 + 0) * LDP + c] = (f16)v.x;
        xs[(n4 + 1) * LDP + c] = (f16)v.y;
        xs[(n4 + 2) * LDP + c] = (f16)v.z;
        xs[(n4 + 3) * LDP + c] = (f16)v.w;
    }
    if (tid < C_) ks[tid] = kvec[b * C_ + tid];

    // ---- wave tiling: 8 waves -> 4 M-tiles x 2 N-groups (64 cols each) ----
    const int wave = tid >> 5;
    const int lane = tid & 31;
    const int ln   = lane & 15;
    const int hi   = lane >> 4;               // half-wave select
    const int mt   = wave & 3;                // output-channel tile
    const int ng   = wave >> 2;               // spatial group (4 subtiles of 16)

    // residual accumulators in WMMA C/D layout, fp32, init = x
    // (GEMM2 accumulates directly into these across both experts)
    v8f facc[4];
    #pragma unroll
    for (int j = 0; j < 4; ++j) {
        int n = hw0 + (ng * 4 + j) * 16 + ln;
        #pragma unroll
        for (int r = 0; r < 8; ++r) {
            int c = mt * 16 + r + hi * 8;
            facc[j][r] = xb[(size_t)c * HW_ + n];
        }
    }

    const int   e0 = sel_idx[b * 2 + 0], e1 = sel_idx[b * 2 + 1];
    const float g0 = sel_w[b * 2 + 0],  g1 = sel_w[b * 2 + 1];

    // warm L2/L0 for the second expert's weight fragments (global_prefetch_b8)
    __builtin_prefetch(W1h + ((size_t)e1 << 12), 0, 0);
    __builtin_prefetch(W2h + ((size_t)e1 << 12), 0, 0);

    for (int s = 0; s < 2; ++s) {
        const int   e  = s ? e1 : e0;
        const float gw = s ? g1 : g0;

        __syncthreads();   // xs ready (s==0) / prior-expert LDS readers done (s==1)

        // stage expert weight fragments (contiguous copy) + folded biases
        const f16* w1p = W1h + ((size_t)e << 12);
        const f16* w2p = W2h + ((size_t)e << 12);
        #pragma unroll
        for (int i = 0; i < 2; ++i) {
            int idx = (((i << 8) + tid) << 3);            // 8 f16 per chunk
            *(v8h*)&w1s[idx] = *(const v8h*)(w1p + idx);
            *(v8h*)&w2s[idx] = *(const v8h*)(w2p + idx);
        }
        if (tid < C_) {
            b1s[tid] = b1g[e * C_ + tid] * ks[tid];       // (W1x+b1)*k = W1x*k + b1*k
            b2s[tid] = gw * b2g[e * C_ + tid];            // gw folded into bias
        }
        __syncthreads();

        // ---------- GEMM1: t = W1 @ x   (bias/scale folded into epilogue) ----------
        v8f hacc[4] = {};
        #pragma unroll
        for (int kk = 0; kk < 2; ++kk) {
            const int afrag = (((kk * 2 + hi) * 64) + mt * 16 + ln) << 4;
            v16h a = *(const v16h*)&w1s[afrag];           // one contiguous 32B read
            const int c0 = kk * 32 + hi * 16;             // B: K 0..15 / 16..31
            #pragma unroll
            for (int j = 0; j < 4; ++j) {
                int n = (ng * 4 + j) * 16 + ln;
                v16h bb = *(const v16h*)&xs[n * LDP + c0];
                hacc[j] = __builtin_amdgcn_wmma_f32_16x16x32_f16(
                    false, a, false, bb, (short)0, hacc[j], false, false);
            }
        }

        // epilogue: h_scaled = gw * gelu(t*k + b1*k), write to LDS B-layout
        const float hsc = 0.5f * gw;
        #pragma unroll
        for (int j = 0; j < 4; ++j) {
            int n = (ng * 4 + j) * 16 + ln;
            #pragma unroll
            for (int r = 0; r < 8; ++r) {
                int c = mt * 16 + r + hi * 8;
                float t = fmaf(hacc[j][r], ks[c], b1s[c]);
                hs[n * LDP + c] = (f16)gelu_scaled(t, hsc);
            }
        }
        __syncthreads();

        // ---------- GEMM2: facc += W2 @ (gw*h)  (direct accumulation) ----------
        #pragma unroll
        for (int kk = 0; kk < 2; ++kk) {
            const int afrag = (((kk * 2 + hi) * 64) + mt * 16 + ln) << 4;
            v16h a = *(const v16h*)&w2s[afrag];
            const int c0 = kk * 32 + hi * 16;
            #pragma unroll
            for (int j = 0; j < 4; ++j) {
                int n = (ng * 4 + j) * 16 + ln;
                v16h bb = *(const v16h*)&hs[n * LDP + c0];
                facc[j] = __builtin_amdgcn_wmma_f32_16x16x32_f16(
                    false, a, false, bb, (short)0, facc[j], false, false);
            }
        }

        // + gw*b2
        #pragma unroll
        for (int j = 0; j < 4; ++j)
            #pragma unroll
            for (int r = 0; r < 8; ++r)
                facc[j][r] += b2s[mt * 16 + r + hi * 8];
    }

    // ---- store out = x + sum w_e*y_e ----
    float* ob = out + (size_t)b * C_ * HW_;
    #pragma unroll
    for (int j = 0; j < 4; ++j) {
        int n = hw0 + (ng * 4 + j) * 16 + ln;
        #pragma unroll
        for (int r = 0; r < 8; ++r) {
            int c = mt * 16 + r + hi * 8;
            ob[(size_t)c * HW_ + n] = facc[j][r];
        }
    }
}

extern "C" void kernel_launch(void* const* d_in, const int* in_sizes, int n_in,
                              void* d_out, int out_size, void* d_ws, size_t ws_size,
                              hipStream_t stream) {
    const float* x  = (const float*)d_in[0];
    const float* k  = (const float*)d_in[1];
    const float* Wg = (const float*)d_in[2];
    const float* bg = (const float*)d_in[3];
    const float* W1 = (const float*)d_in[4];
    const float* b1 = (const float*)d_in[5];
    const float* W2 = (const float*)d_in[6];
    const float* b2 = (const float*)d_in[7];
    float* out = (float*)d_out;

    char* ws = (char*)d_ws;
    float* pooled = (float*)ws;                                   //  4 KB
    float* selw   = (float*)(ws + 4096);                          //  128 B
    int*   seli   = (int*)  (ws + 4096 + 256);                    //  128 B
    f16*   W1h    = (f16*)  (ws + 8192);                          // 64 KB (fragment order)
    f16*   W2h    = (f16*)  (ws + 8192 + E_ * C_ * C_ * 2);       // 64 KB (fragment order)

    pool_kernel<<<B_ * C_, 256, 0, stream>>>(x, pooled);
    gate_kernel<<<1, 256, 0, stream>>>(pooled, Wg, bg, W1, W2, W1h, W2h, seli, selw);
    moe_expert_kernel<<<B_ * (HW_ / NT), 256, 0, stream>>>(
        x, k, b1, b2, W1h, W2h, seli, selw, out);
}
